// mIHC_GIN_surv_83494164234289
// MI455X (gfx1250) — matrix operand order repaired and checked by reading, not compiled, not run on hardware
//
#include <hip/hip_runtime.h>
#include <hip/hip_bf16.h>

// ---------------------------------------------------------------------------
// mIHC GIN-surv pipeline for gfx1250 (MI455X), wave32 + WMMA f16->f32.
// ---------------------------------------------------------------------------

typedef __attribute__((ext_vector_type(16))) _Float16 v16h;
typedef __attribute__((ext_vector_type(8)))  _Float16 v8h;
typedef __attribute__((ext_vector_type(8)))  float    v8f;

#define NB      8
#define NPER1   10000
#define NPER2   6000
#define NODES1  80000   /* NB*NPER1 */
#define NODES2  48000   /* NB*KEEP1 */
#define KEEP1   6000
#define KEEP2   3600
#define CCH     128
#define FDIM    64
#define NEDGE   640000
#define NBINS   2048

// ------------------------------ helpers ------------------------------------

__device__ __forceinline__ void bin_of(float s, int& bin, int& sub) {
    float f = (s + 1.0f) * 1024.0f;          // tanh in (-1,1) -> [0,2048)
    int b = (int)f;
    b = b < 0 ? 0 : (b > NBINS - 1 ? NBINS - 1 : b);
    float fr = (f - (float)b) * 2048.0f;
    int su = (int)fr;
    su = su < 0 ? 0 : (su > NBINS - 1 ? NBINS - 1 : su);
    bin = b; sub = su;
}

// transpose W[K,N] f32 -> WT[N,K] f16
__global__ void k_wT(const float* __restrict__ W, _Float16* __restrict__ WT,
                     int K, int N) {
    int i = blockIdx.x * blockDim.x + threadIdx.x;
    if (i >= K * N) return;
    int k = i / N, n = i % N;
    WT[n * K + k] = (_Float16)W[i];
}

// o = f16(a [+ b])
__global__ void k_cvt(const float* __restrict__ a, const float* __restrict__ b,
                      _Float16* __restrict__ o, int n) {
    int i = blockIdx.x * blockDim.x + threadIdx.x;
    if (i >= n) return;
    float v = a[i];
    if (b) v += b[i];
    o[i] = (_Float16)v;
}

// ------------------------------ WMMA GEMM ----------------------------------
// C[M,128] = act(A16[M,K] @ (BT16[128,K])^T + bias), M % 128 == 0, K % 32 == 0
template <bool RELU>
__global__ __launch_bounds__(256) void k_gemm(
    const _Float16* __restrict__ A, const _Float16* __restrict__ BT,
    const float* __restrict__ bias, float* __restrict__ C, int M, int K) {
    const int lane = threadIdx.x & 31;
    const int wave = threadIdx.x >> 5;
    const int lhi  = lane >> 4;                 // 0 for lanes 0-15, 1 for 16-31
    const int l15  = lane & 15;
    const int mbase = blockIdx.x * 128 + wave * 16;
    const int row  = mbase + l15;

    v8f acc[8];
#pragma unroll
    for (int t = 0; t < 8; ++t)
#pragma unroll
        for (int j = 0; j < 8; ++j) acc[t][j] = 0.0f;

    for (int kb = 0; kb < K; kb += 32) {
        // A fragment: halves 0..15 -> K = kb + 16*lhi + h (contiguous)
        const _Float16* pA = A + (size_t)row * K + kb + (lhi ? 16 : 0);
        v8h a0 = *(const v8h*)(pA);
        v8h a1 = *(const v8h*)(pA + 8);
        v16h af;
#pragma unroll
        for (int i = 0; i < 8; ++i) { af[i] = a0[i]; af[8 + i] = a1[i]; }
#pragma unroll
        for (int t = 0; t < 8; ++t) {
            // B fragment: column n = t*16 + l15, 16 contiguous K from kb+16*lhi
            const _Float16* pB = BT + (size_t)(t * 16 + l15) * K + kb + (lhi ? 16 : 0);
            v16h bf = *(const v16h*)pB;
            acc[t] = __builtin_amdgcn_wmma_f32_16x16x32_f16(
                false, af, false, bf, (short)0, acc[t], false, false);
        }
    }
    // D: lane<16: N=l15, M=r ; lane>=16: N=l15, M=8+r
#pragma unroll
    for (int t = 0; t < 8; ++t) {
        const int col = t * 16 + l15;
        const float bb = bias ? bias[col] : 0.0f;
#pragma unroll
        for (int r = 0; r < 8; ++r) {
            float v = acc[t][r] + bb;
            if (RELU) v = v > 0.0f ? v : 0.0f;
            C[(size_t)(mbase + r + 8 * lhi) * 128 + col] = v;
        }
    }
}

// ------------------------- edge segment-sum --------------------------------
// agg[dst] += h[src], one wave per edge, 4 channels per lane, -1 = masked
__global__ void k_segsum(const float* __restrict__ h, const int* __restrict__ src,
                         const int* __restrict__ dst, float* __restrict__ agg,
                         int nE) {
    int t = blockIdx.x * blockDim.x + threadIdx.x;
    int e = t >> 5;
    if (e >= nE) return;
    int s = src[e], d = dst[e];
    if (s < 0 || d < 0) return;
    int c0 = (t & 31) * 4;
    const float* ph = h + (size_t)s * CCH + c0;
    float* pa = agg + (size_t)d * CCH + c0;
#pragma unroll
    for (int j = 0; j < 4; ++j) atomicAdd(pa + j, ph[j]);
}

// ----------------------- SAG pooling: score --------------------------------
// score = tanh(agg.Wrel + brel + h.Wroot) ; one wave per node
__global__ void k_score(const float* __restrict__ h, const float* __restrict__ agg,
                        const float* __restrict__ Wrel, const float* __restrict__ Wroot,
                        const float* __restrict__ brel, float* __restrict__ score,
                        int M) {
    int t = blockIdx.x * blockDim.x + threadIdx.x;
    int node = t >> 5, lane = t & 31;
    if (node >= M) return;
    float s = 0.0f;
#pragma unroll
    for (int j = 0; j < 4; ++j) {
        int c = lane * 4 + j;
        s += agg[(size_t)node * CCH + c] * Wrel[c] + h[(size_t)node * CCH + c] * Wroot[c];
    }
#pragma unroll
    for (int off = 16; off > 0; off >>= 1) s += __shfl_down(s, off, 32);
    if (lane == 0) score[node] = tanhf(s + brel[0]);
}

// -------------------- top-K via 2-level histogram ---------------------------
__global__ void k_hist1(const float* __restrict__ score, int* __restrict__ hist,
                        int M, int nPer) {
    int i = blockIdx.x * blockDim.x + threadIdx.x;
    if (i >= M) return;
    int b, su; bin_of(score[i], b, su);
    atomicAdd(&hist[(i / nPer) * NBINS + b], 1);
}

__global__ void k_hist2(const float* __restrict__ score, const int* __restrict__ thrB,
                        int* __restrict__ hist, int M, int nPer) {
    int i = blockIdx.x * blockDim.x + threadIdx.x;
    if (i >= M) return;
    int g = i / nPer;
    int b, su; bin_of(score[i], b, su);
    if (b != thrB[g]) return;
    atomicAdd(&hist[g * NBINS + su], 1);
}

// per-graph serial scan from the top to find threshold bin + residual need
__global__ void k_thr(const int* __restrict__ hist, int kimm,
                      const int* __restrict__ kptr, int* __restrict__ thrBin,
                      int* __restrict__ thrNeed) {
    int g = threadIdx.x;
    if (g >= NB) return;
    int Ksel = kptr ? kptr[g] : kimm;
    int cum = 0;
    for (int b = NBINS - 1; b >= 0; --b) {
        int c = hist[g * NBINS + b];
        if (cum + c >= Ksel) { thrBin[g] = b; thrNeed[g] = Ksel - cum; return; }
        cum += c;
    }
    thrBin[g] = 0; thrNeed[g] = Ksel - cum;
}

__global__ void k_select(const float* __restrict__ score, const int* __restrict__ tB1,
                         const int* __restrict__ tB2, const int* __restrict__ tN2,
                         int* __restrict__ cntTie, int* __restrict__ cntSel,
                         int* __restrict__ new_id, int M, int nPer, int Kkeep) {
    int i = blockIdx.x * blockDim.x + threadIdx.x;
    if (i >= M) return;
    int g = i / nPer;
    int b, su; bin_of(score[i], b, su);
    bool sel = false;
    if (b > tB1[g]) sel = true;
    else if (b == tB1[g]) {
        if (su > tB2[g]) sel = true;
        else if (su == tB2[g]) sel = atomicAdd(&cntTie[g], 1) < tN2[g];
    }
    if (sel) {
        int pos = atomicAdd(&cntSel[g], 1);
        new_id[i] = g * Kkeep + pos;
    } else new_id[i] = -1;
}

// gather selected nodes: xp[new_id] = h[i]*score[i]; accumulate per-graph mean
__global__ void k_gather(const float* __restrict__ h, const float* __restrict__ score,
                         const int* __restrict__ new_id, float* __restrict__ xp,
                         float* __restrict__ outMean, int M, int nPer, float invK) {
    int t = blockIdx.x * blockDim.x + threadIdx.x;
    int i = t >> 5;
    if (i >= M) return;
    int nid = new_id[i];
    if (nid < 0) return;
    int g = i / nPer;
    int c0 = (t & 31) * 4;
    float sc = score[i];
#pragma unroll
    for (int j = 0; j < 4; ++j) {
        float v = h[(size_t)i * CCH + c0 + j] * sc;
        if (xp) xp[(size_t)nid * CCH + c0 + j] = v;
        atomicAdd(&outMean[g * CCH + c0 + j], v * invK);
    }
}

__global__ void k_remap(const int* __restrict__ src, const int* __restrict__ dst,
                        const int* __restrict__ new_id, int* __restrict__ src2,
                        int* __restrict__ dst2, int nE) {
    int e = blockIdx.x * blockDim.x + threadIdx.x;
    if (e >= nE) return;
    int s = new_id[src[e]], d = new_id[dst[e]];
    if (s < 0 || d < 0) { s = -1; d = -1; }
    src2[e] = s; dst2[e] = d;
}

// --------------------------- head MLP (tiny) --------------------------------
__global__ __launch_bounds__(256) void k_head(
    const float* __restrict__ out1, const float* __restrict__ out2,
    const float* __restrict__ Wg, const float* __restrict__ bg,
    const float* __restrict__ Wr1, const float* __restrict__ br1,
    const float* __restrict__ g1, const float* __restrict__ be1,
    const float* __restrict__ Wr2, const float* __restrict__ br2,
    const float* __restrict__ g2, const float* __restrict__ be2,
    const float* __restrict__ Wout, float* __restrict__ out) {
    __shared__ float sg[8 * 128];
    __shared__ float sr1[8 * 64];
    __shared__ float sr2[8 * 32];
    const float inv = 0.99999499993750047f;   // 1/sqrt(1+1e-5)
    int t = threadIdx.x;
    for (int idx = t; idx < 8 * 128; idx += 256) {
        int r = idx >> 7, c = idx & 127;
        float acc = bg[c];
        for (int k = 0; k < 128; ++k) acc += out1[r * 128 + k] * Wg[k * 128 + c];
        for (int k = 0; k < 128; ++k) acc += out2[r * 128 + k] * Wg[(128 + k) * 128 + c];
        sg[idx] = acc;
    }
    __syncthreads();
    for (int idx = t; idx < 8 * 64; idx += 256) {
        int r = idx >> 6, c = idx & 63;
        float acc = br1[c];
        for (int k = 0; k < 128; ++k) acc += sg[r * 128 + k] * Wr1[k * 64 + c];
        acc = acc * inv * g1[c] + be1[c];
        sr1[idx] = acc > 0.0f ? acc : 0.0f;
    }
    __syncthreads();
    for (int idx = t; idx < 8 * 32; idx += 256) {
        int r = idx >> 5, c = idx & 31;
        float acc = br2[c];
        for (int k = 0; k < 64; ++k) acc += sr1[r * 64 + k] * Wr2[k * 32 + c];
        acc = acc * inv * g2[c] + be2[c];
        sr2[idx] = acc > 0.0f ? acc : 0.0f;
    }
    __syncthreads();
    for (int idx = t; idx < 32; idx += 256) {
        int r = idx >> 2, c = idx & 3;
        float acc = 0.0f;
        for (int k = 0; k < 32; ++k) acc += sr2[r * 32 + k] * Wout[k * 4 + c];
        out[idx] = acc;
    }
}

// ------------------------------- launcher -----------------------------------

extern "C" void kernel_launch(void* const* d_in, const int* in_sizes, int n_in,
                              void* d_out, int out_size, void* d_ws, size_t ws_size,
                              hipStream_t stream) {
    (void)in_sizes; (void)n_in; (void)out_size; (void)ws_size;

    const float* x     = (const float*)d_in[0];
    const int*   edge  = (const int*)d_in[1];
    // d_in[2] = batch (implied by node layout, unused)
    const float* W0    = (const float*)d_in[3];
    const float* b0    = (const float*)d_in[4];
    const float* W1a   = (const float*)d_in[5];
    const float* b1a   = (const float*)d_in[6];
    const float* W1b   = (const float*)d_in[7];
    const float* b1b   = (const float*)d_in[8];
    const float* Wrel1 = (const float*)d_in[9];
    const float* brel1 = (const float*)d_in[10];
    const float* Wroot1= (const float*)d_in[11];
    const float* W2a   = (const float*)d_in[12];
    const float* b2a   = (const float*)d_in[13];
    const float* W2b   = (const float*)d_in[14];
    const float* b2b   = (const float*)d_in[15];
    const float* Wrel2 = (const float*)d_in[16];
    const float* brel2 = (const float*)d_in[17];
    const float* Wroot2= (const float*)d_in[18];
    const float* Wg    = (const float*)d_in[19];
    const float* bg    = (const float*)d_in[20];
    const float* Wr1   = (const float*)d_in[21];
    const float* br1   = (const float*)d_in[22];
    const float* g1    = (const float*)d_in[23];
    const float* be1   = (const float*)d_in[24];
    const float* Wr2   = (const float*)d_in[25];
    const float* br2   = (const float*)d_in[26];
    const float* g2    = (const float*)d_in[27];
    const float* be2   = (const float*)d_in[28];
    const float* Wout  = (const float*)d_in[29];
    const int* srcIdx = edge;
    const int* dstIdx = edge + NEDGE;

    // -------- workspace carve (aliased where lifetimes permit) --------
    char* base = (char*)d_ws;
    size_t off = 0;
    auto carve = [&](size_t bytes) -> char* {
        char* r = base + off;
        off += (bytes + 255) & ~(size_t)255;
        return r;
    };
    float*     h0   = (float*)carve((size_t)NODES1 * CCH * 4);   // reused as h2
    float*     agg  = (float*)carve((size_t)NODES1 * CCH * 4);
    _Float16*  a16  = (_Float16*)carve((size_t)NODES1 * CCH * 2);
    float*     tmp  = (float*)carve((size_t)NODES1 * CCH * 4);   // reused as xp1
    float*     h1   = (float*)carve((size_t)NODES1 * CCH * 4);
    float*     score= (float*)carve((size_t)NODES1 * 4);
    int*       hist1= (int*)carve(NB * NBINS * 4);
    int*       hist2= (int*)carve(NB * NBINS * 4);
    int*       thrB1= (int*)carve(NB * 4);
    int*       thrN1= (int*)carve(NB * 4);
    int*       thrB2= (int*)carve(NB * 4);
    int*       thrN2= (int*)carve(NB * 4);
    int*       cnt  = (int*)carve(2 * NB * 4);   // [cntTie | cntSel]
    int*       newid= (int*)carve((size_t)NODES1 * 4);
    int*       src2 = (int*)carve((size_t)NEDGE * 4);
    int*       dst2 = (int*)carve((size_t)NEDGE * 4);
    _Float16*  w0t  = (_Float16*)carve(FDIM * CCH * 2);
    _Float16*  w1at = (_Float16*)carve(CCH * CCH * 2);
    _Float16*  w1bt = (_Float16*)carve(CCH * CCH * 2);
    _Float16*  w2at = (_Float16*)carve(CCH * CCH * 2);
    _Float16*  w2bt = (_Float16*)carve(CCH * CCH * 2);
    float*     out1 = (float*)carve(NB * CCH * 4);
    float*     out2 = (float*)carve(NB * CCH * 4);
    float*     h2   = h0;    // alias: h0 dead after GIN1 A-conversion
    float*     xp1  = tmp;   // alias: tmp dead before xp1 written, and vice versa
    int* cntTie = cnt; int* cntSel = cnt + NB;

    const dim3 B256(256);
    const int edgeGrid = (NEDGE * 32) / 256;     // exact: 80000

    // -------- weights -> f16 transposed [N][K] --------
    k_wT<<<(FDIM * CCH + 255) / 256, B256, 0, stream>>>(W0, w0t, FDIM, CCH);
    k_wT<<<(CCH * CCH + 255) / 256, B256, 0, stream>>>(W1a, w1at, CCH, CCH);
    k_wT<<<(CCH * CCH + 255) / 256, B256, 0, stream>>>(W1b, w1bt, CCH, CCH);
    k_wT<<<(CCH * CCH + 255) / 256, B256, 0, stream>>>(W2a, w2at, CCH, CCH);
    k_wT<<<(CCH * CCH + 255) / 256, B256, 0, stream>>>(W2b, w2bt, CCH, CCH);

    // -------- stage 0: h0 = relu(x @ W0 + b0) --------
    k_cvt<<<(NODES1 * FDIM + 255) / 256, B256, 0, stream>>>(x, nullptr, a16, NODES1 * FDIM);
    k_gemm<true><<<NODES1 / 128, B256, 0, stream>>>(a16, w0t, b0, h0, NODES1, FDIM);

    // -------- GIN layer 1 --------
    hipMemsetAsync(agg, 0, (size_t)NODES1 * CCH * 4, stream);
    k_segsum<<<edgeGrid, B256, 0, stream>>>(h0, srcIdx, dstIdx, agg, NEDGE);
    k_cvt<<<(NODES1 * CCH + 255) / 256, B256, 0, stream>>>(h0, agg, a16, NODES1 * CCH);
    k_gemm<true><<<NODES1 / 128, B256, 0, stream>>>(a16, w1at, b1a, tmp, NODES1, CCH);
    k_cvt<<<(NODES1 * CCH + 255) / 256, B256, 0, stream>>>(tmp, nullptr, a16, NODES1 * CCH);
    k_gemm<false><<<NODES1 / 128, B256, 0, stream>>>(a16, w1bt, b1b, h1, NODES1, CCH);

    // -------- SAG pool 1 --------
    hipMemsetAsync(agg, 0, (size_t)NODES1 * CCH * 4, stream);
    k_segsum<<<edgeGrid, B256, 0, stream>>>(h1, srcIdx, dstIdx, agg, NEDGE);
    k_score<<<(NODES1 * 32) / 256, B256, 0, stream>>>(h1, agg, Wrel1, Wroot1, brel1, score, NODES1);
    hipMemsetAsync(hist1, 0, NB * NBINS * 4, stream);
    k_hist1<<<(NODES1 + 255) / 256, B256, 0, stream>>>(score, hist1, NODES1, NPER1);
    k_thr<<<1, NB, 0, stream>>>(hist1, KEEP1, nullptr, thrB1, thrN1);
    hipMemsetAsync(hist2, 0, NB * NBINS * 4, stream);
    k_hist2<<<(NODES1 + 255) / 256, B256, 0, stream>>>(score, thrB1, hist2, NODES1, NPER1);
    k_thr<<<1, NB, 0, stream>>>(hist2, 0, thrN1, thrB2, thrN2);
    hipMemsetAsync(cnt, 0, 2 * NB * 4, stream);
    k_select<<<(NODES1 + 255) / 256, B256, 0, stream>>>(score, thrB1, thrB2, thrN2,
                                                        cntTie, cntSel, newid,
                                                        NODES1, NPER1, KEEP1);
    hipMemsetAsync(out1, 0, NB * CCH * 4, stream);
    k_gather<<<(NODES1 * 32) / 256, B256, 0, stream>>>(h1, score, newid, xp1, out1,
                                                       NODES1, NPER1, 1.0f / KEEP1);
    k_remap<<<(NEDGE + 255) / 256, B256, 0, stream>>>(srcIdx, dstIdx, newid, src2, dst2, NEDGE);

    // -------- GIN layer 2 --------
    hipMemsetAsync(agg, 0, (size_t)NODES2 * CCH * 4, stream);
    k_segsum<<<edgeGrid, B256, 0, stream>>>(xp1, src2, dst2, agg, NEDGE);
    k_cvt<<<(NODES2 * CCH + 255) / 256, B256, 0, stream>>>(xp1, agg, a16, NODES2 * CCH);
    k_gemm<true><<<NODES2 / 128, B256, 0, stream>>>(a16, w2at, b2a, tmp, NODES2, CCH);
    k_cvt<<<(NODES2 * CCH + 255) / 256, B256, 0, stream>>>(tmp, nullptr, a16, NODES2 * CCH);
    k_gemm<false><<<NODES2 / 128, B256, 0, stream>>>(a16, w2bt, b2b, h2, NODES2, CCH);

    // -------- SAG pool 2 (edges discarded afterwards; only mean needed) ------
    hipMemsetAsync(agg, 0, (size_t)NODES2 * CCH * 4, stream);
    k_segsum<<<edgeGrid, B256, 0, stream>>>(h2, src2, dst2, agg, NEDGE);
    k_score<<<(NODES2 * 32) / 256, B256, 0, stream>>>(h2, agg, Wrel2, Wroot2, brel2, score, NODES2);
    hipMemsetAsync(hist1, 0, NB * NBINS * 4, stream);
    k_hist1<<<(NODES2 + 255) / 256, B256, 0, stream>>>(score, hist1, NODES2, NPER2);
    k_thr<<<1, NB, 0, stream>>>(hist1, KEEP2, nullptr, thrB1, thrN1);
    hipMemsetAsync(hist2, 0, NB * NBINS * 4, stream);
    k_hist2<<<(NODES2 + 255) / 256, B256, 0, stream>>>(score, thrB1, hist2, NODES2, NPER2);
    k_thr<<<1, NB, 0, stream>>>(hist2, 0, thrN1, thrB2, thrN2);
    hipMemsetAsync(cnt, 0, 2 * NB * 4, stream);
    k_select<<<(NODES2 + 255) / 256, B256, 0, stream>>>(score, thrB1, thrB2, thrN2,
                                                        cntTie, cntSel, newid,
                                                        NODES2, NPER2, KEEP2);
    hipMemsetAsync(out2, 0, NB * CCH * 4, stream);
    k_gather<<<(NODES2 * 32) / 256, B256, 0, stream>>>(h2, score, newid, nullptr, out2,
                                                       NODES2, NPER2, 1.0f / KEEP2);

    // -------- head --------
    k_head<<<1, B256, 0, stream>>>(out1, out2, Wg, bg, Wr1, br1, g1, be1,
                                   Wr2, br2, g2, be2, Wout, (float*)d_out);
}
